// TemporalGNN_80023830659660
// MI455X (gfx1250) — compile-verified, matrix operand order checked by reading, not run on hardware
//
#include <hip/hip_runtime.h>
#include <hip/hip_bf16.h>

// ---------------------------------------------------------------------------
// Types / WMMA helpers (gfx1250, wave32)
// ---------------------------------------------------------------------------
typedef __attribute__((ext_vector_type(16))) __bf16 v16bf;
typedef __attribute__((ext_vector_type(8)))  float  v8f;

#define WMMA_BF16(a, b, c) \
  __builtin_amdgcn_wmma_f32_16x16x32_bf16(false, (a), false, (b), (short)0, (c), false, false)

static __device__ __forceinline__ __bf16 f2bf(float f) { return (__bf16)f; }

static __device__ __forceinline__ unsigned short bfbits(float f) {
  union { __bf16 b; unsigned short s; } u;
  u.b = f2bf(f);
  return u.s;
}

static __device__ __forceinline__ float sigm(float v) { return 1.0f / (1.0f + __expf(-v)); }

// Build an A-fragment (16x32 bf16, MxK) from an LDS fp32 tile S[16][132].
// Lane L: row = L&15; elems 0..7 -> K = kb..kb+7, elems 8..15 -> K = kb+16..kb+23,
// where kb = 32*ks + (L>=16 ? 8 : 0).   (ISA 7.12.2, 16-bit A 16x32 layout)
static __device__ __forceinline__ v16bf makeA(const float* S, int lane, int ks) {
  const float* p = S + (lane & 15) * 132 + ks * 32 + ((lane & 16) ? 8 : 0);
  v16bf a;
#pragma unroll
  for (int e = 0; e < 8; ++e) a[e] = f2bf(p[e]);
#pragma unroll
  for (int e = 0; e < 8; ++e) a[8 + e] = f2bf(p[16 + e]);
  return a;
}

// Load a pre-packed B-fragment: lane's 16 bf16 stored as 8 consecutive uints.
// Works for both global (ws) and LDS staged copies (generic pointer).
static __device__ __forceinline__ v16bf ldB(const unsigned int* pk, int frag, int lane) {
  const uint4* p = (const uint4*)(pk + ((size_t)(frag * 32 + lane)) * 8);
  uint4 a = p[0], b = p[1];
  union { unsigned int u[8]; v16bf v; } t;
  t.u[0] = a.x; t.u[1] = a.y; t.u[2] = a.z; t.u[3] = a.w;
  t.u[4] = b.x; t.u[5] = b.y; t.u[6] = b.z; t.u[7] = b.w;
  return t.v;
}

// ---------------------------------------------------------------------------
// Pack W[Ngate,K] (fp32, row-major) into B-fragment-major bf16 (B = W^T).
// ---------------------------------------------------------------------------
__global__ void pack_w_kernel(const float* __restrict__ W, unsigned int* __restrict__ out, int K) {
  int frag = blockIdx.x;
  int KS = K >> 5;
  int nt = frag / KS, ks = frag - nt * KS;
  int lane = threadIdx.x;
  int n = nt * 16 + (lane & 15);
  int kb = ks * 32 + ((lane & 16) ? 8 : 0);
  const float* row = W + (size_t)n * K;
  unsigned int* dst = out + ((size_t)(frag * 32 + lane)) * 8;
#pragma unroll
  for (int j = 0; j < 8; ++j) {
    int k0 = kb + 2 * j + ((j >= 4) ? 8 : 0);
    unsigned int lo = bfbits(row[k0]);
    unsigned int hi = bfbits(row[k0 + 1]);
    dst[j] = lo | (hi << 16);
  }
}

// ---------------------------------------------------------------------------
// Graph build: per batch, Gram = Xf Xf^T (K=32768) via bf16 WMMA, then cosine
// normalize, top-3 per row, clamp, symmetrize, self-loops -> wg[b][64][64].
// For a Gram matrix with this fragment layout, A-frag(tile) == B-frag(tile).
// ---------------------------------------------------------------------------
static __device__ __forceinline__ v16bf ldX(const float* x, int b, int tile, int lane, int ks) {
  size_t row = (size_t)(b * 64 + tile * 16 + (lane & 15));
  const float* p = x + row * 32768 + ks * 32 + ((lane & 16) ? 8 : 0);
  const float4* q = (const float4*)p;
  float4 u0 = q[0], u1 = q[1], u2 = q[4], u3 = q[5];
  v16bf a;
  a[0] = f2bf(u0.x);  a[1] = f2bf(u0.y);  a[2] = f2bf(u0.z);  a[3] = f2bf(u0.w);
  a[4] = f2bf(u1.x);  a[5] = f2bf(u1.y);  a[6] = f2bf(u1.z);  a[7] = f2bf(u1.w);
  a[8] = f2bf(u2.x);  a[9] = f2bf(u2.y);  a[10] = f2bf(u2.z); a[11] = f2bf(u2.w);
  a[12] = f2bf(u3.x); a[13] = f2bf(u3.y); a[14] = f2bf(u3.z); a[15] = f2bf(u3.w);
  return a;
}

__global__ void __launch_bounds__(256) graph_build_kernel(const float* __restrict__ x,
                                                          float* __restrict__ wg) {
  __shared__ float G[64 * 66];
  __shared__ float ADJ[64 * 64];
  __shared__ float NRM[64];
  int b = blockIdx.x, tid = threadIdx.x, lane = tid & 31, w = tid >> 5;
  int t0 = w, t1 = w + 8;
  int m0 = t0 >> 2, n0 = t0 & 3, m1 = t1 >> 2, n1 = t1 & 3;
  v8f c0, c1;
#pragma unroll
  for (int r = 0; r < 8; ++r) { c0[r] = 0.f; c1[r] = 0.f; }
  for (int ks = 0; ks < 1024; ++ks) {
    v16bf am0 = ldX(x, b, m0, lane, ks);
    v16bf bn0 = ldX(x, b, n0, lane, ks);
    v16bf am1 = ldX(x, b, m1, lane, ks);
    v16bf bn1 = ldX(x, b, n1, lane, ks);
    c0 = WMMA_BF16(am0, bn0, c0);
    c1 = WMMA_BF16(am1, bn1, c1);
  }
  int mb = (lane & 16) ? 8 : 0, nn = lane & 15;
#pragma unroll
  for (int r = 0; r < 8; ++r) G[(16 * m0 + r + mb) * 66 + 16 * n0 + nn] = c0[r];
#pragma unroll
  for (int r = 0; r < 8; ++r) G[(16 * m1 + r + mb) * 66 + 16 * n1 + nn] = c1[r];
  __syncthreads();
  if (tid < 64) NRM[tid] = rsqrtf(fmaxf(G[tid * 66 + tid], 1e-30f));
  for (int e = tid; e < 4096; e += 256) ADJ[e] = 0.f;
  __syncthreads();
  if (tid < 64) {
    int n = tid;
    float rn = NRM[n];
    float v0 = -1e30f, v1 = -1e30f, v2 = -1e30f;
    int i0 = 0, i1 = 0, i2 = 0;
    for (int m = 0; m < 64; ++m) {
      float s = G[n * 66 + m] * rn * NRM[m];
      if (s > v0)      { v2 = v1; i2 = i1; v1 = v0; i1 = i0; v0 = s; i0 = m; }
      else if (s > v1) { v2 = v1; i2 = i1; v1 = s; i1 = m; }
      else if (s > v2) { v2 = s; i2 = m; }
    }
    ADJ[n * 64 + i0] = v0; ADJ[n * 64 + i1] = v1; ADJ[n * 64 + i2] = v2;
  }
  __syncthreads();
  for (int e = tid; e < 4096; e += 256) {
    int n = e >> 6, m = e & 63;
    float a = fmaxf(ADJ[n * 64 + m], 0.f);
    float at = fmaxf(ADJ[m * 64 + n], 0.f);
    float wv = 0.5f * (a + at);
    if (n == m) wv = 1.0f;
    wg[(size_t)b * 4096 + e] = wv;
  }
}

// ---------------------------------------------------------------------------
// Fused 2-layer GRU scan + mean pool.  64 WGs x 16 rows; wave w owns hidden
// cols 16w..16w+15 and gate N-tiles {w, w+8, w+16} so r/z/n stay in-register.
// CDNA5 320KB LDS: Whh0/Wih1/Whh1 staged in LDS (loop-invariant).
// Wih0 fragments pinned in registers (12 x v16bf per wave, loaded once).
// x_t is software-pipelined: prefetch x_{t+1} into registers at step start,
// commit to the (single) Xt LDS buffer once all waves are done reading x_t.
// Steady-state inner loop: only one pipelined global access (x prefetch).
// ---------------------------------------------------------------------------
#define WFRAG_UINTS (96 * 32 * 8)  // 24576 uints = 96KB per packed matrix

__global__ void __launch_bounds__(256) gru_scan_kernel(
    const float* __restrict__ x,
    const unsigned int* __restrict__ pWih0, const unsigned int* __restrict__ pWhh0,
    const unsigned int* __restrict__ pWih1, const unsigned int* __restrict__ pWhh1,
    const float* __restrict__ bih0, const float* __restrict__ bhh0,
    const float* __restrict__ bih1, const float* __restrict__ bhh1,
    float* __restrict__ hmean) {
  __shared__ float Xt[16 * 132];
  __shared__ float H0[16 * 132];
  __shared__ float H1[16 * 132];
  __shared__ unsigned int sWhh0[WFRAG_UINTS];
  __shared__ unsigned int sWih1[WFRAG_UINTS];
  __shared__ unsigned int sWhh1[WFRAG_UINTS];
  const int tid = threadIdx.x;
  const int lane = tid & 31;
  const int w = tid >> 5;
  const int row0 = blockIdx.x * 16;

  // Stage loop-invariant weights into LDS once (128-bit copies).
  {
    const uint4* s0 = (const uint4*)pWhh0;
    const uint4* s1 = (const uint4*)pWih1;
    const uint4* s2 = (const uint4*)pWhh1;
    uint4* d0 = (uint4*)sWhh0;
    uint4* d1 = (uint4*)sWih1;
    uint4* d2 = (uint4*)sWhh1;
    for (int i = tid; i < WFRAG_UINTS / 4; i += 256) {
      d0[i] = s0[i];
      d1[i] = s1[i];
      d2[i] = s2[i];
    }
  }
  for (int i = tid; i < 16 * 132; i += 256) { H0[i] = 0.f; H1[i] = 0.f; }

  // Pin the Wih0 B-fragments this wave needs in registers (loop-invariant).
  v16bf Bx0[12];
#pragma unroll
  for (int g = 0; g < 3; ++g)
#pragma unroll
    for (int ks = 0; ks < 4; ++ks)
      Bx0[g * 4 + ks] = ldB(pWih0, (w + 8 * g) * 4 + ks, lane);

  const int j = 16 * w + (lane & 15);  // hidden column owned by (wave,lane)
  const int mbase = (lane & 16) ? 8 : 0;

  float bx0[3], bh0[3], bx1[3], bh1[3];
#pragma unroll
  for (int g = 0; g < 3; ++g) {
    bx0[g] = bih0[128 * g + j]; bh0[g] = bhh0[128 * g + j];
    bx1[g] = bih1[128 * g + j]; bh1[g] = bhh1[128 * g + j];
  }
  float acc[8];
#pragma unroll
  for (int r = 0; r < 8; ++r) acc[r] = 0.f;

  // Per-thread x staging addresses (each thread moves 2 float4 per step).
  const int m0_ = tid >> 5,          d40_ = tid & 31;
  const int m1_ = (tid + 256) >> 5,  d41_ = (tid + 256) & 31;
  const float* xp0 = &x[((size_t)(row0 + m0_) * 256) * 128 + d40_ * 4];
  const float* xp1 = &x[((size_t)(row0 + m1_) * 256) * 128 + d41_ * 4];
  float* xs0 = &Xt[m0_ * 132 + d40_ * 4];
  float* xs1 = &Xt[m1_ * 132 + d41_ * 4];

  // Prologue: stage x_0.
  float4 xn0 = *(const float4*)xp0;
  float4 xn1 = *(const float4*)xp1;
  *(float4*)xs0 = xn0;
  *(float4*)xs1 = xn1;
  __syncthreads();

  for (int t = 0; t < 256; ++t) {
    // Fire-and-forget prefetch of x_{t+1} into registers (latency hidden
    // behind layer-0 WMMA; waits auto-inserted at the commit below).
    int tn = (t < 255) ? (t + 1) : 255;
    xn0 = *(const float4*)&xp0[(size_t)tn * 128];
    xn1 = *(const float4*)&xp1[(size_t)tn * 128];

    // ---- layer 0: Gx = x_t@Wih0^T (reg-pinned B), Gh = h0@Whh0^T (LDS) ----
    v16bf Ax[4], Ah[4];
#pragma unroll
    for (int ks = 0; ks < 4; ++ks) { Ax[ks] = makeA(Xt, lane, ks); Ah[ks] = makeA(H0, lane, ks); }
    v8f cx[3], ch[3];
#pragma unroll
    for (int g = 0; g < 3; ++g) {
#pragma unroll
      for (int r = 0; r < 8; ++r) { cx[g][r] = bx0[g]; ch[g][r] = bh0[g]; }
      int nt = w + 8 * g;
#pragma unroll
      for (int ks = 0; ks < 4; ++ks) {
        cx[g] = WMMA_BF16(Ax[ks], Bx0[g * 4 + ks], cx[g]);
        ch[g] = WMMA_BF16(Ah[ks], ldB(sWhh0, nt * 4 + ks, lane), ch[g]);
      }
    }
    __syncthreads();  // all waves done reading Xt (x_t) and H0
#pragma unroll
    for (int r = 0; r < 8; ++r) {
      int m = r + mbase;
      float hp = H0[m * 132 + j];
      float rr = sigm(cx[0][r] + ch[0][r]);
      float zz = sigm(cx[1][r] + ch[1][r]);
      float nn = tanhf(cx[2][r] + rr * ch[2][r]);
      H0[m * 132 + j] = (1.f - zz) * nn + zz * hp;
    }
    // Commit prefetched x_{t+1} into Xt (x_t dead after the barrier above).
    *(float4*)xs0 = xn0;
    *(float4*)xs1 = xn1;
    __syncthreads();

    // ---- layer 1 (consumes layer-0 h_t immediately; all weights in LDS) ----
#pragma unroll
    for (int ks = 0; ks < 4; ++ks) { Ax[ks] = makeA(H0, lane, ks); Ah[ks] = makeA(H1, lane, ks); }
#pragma unroll
    for (int g = 0; g < 3; ++g) {
#pragma unroll
      for (int r = 0; r < 8; ++r) { cx[g][r] = bx1[g]; ch[g][r] = bh1[g]; }
      int nt = w + 8 * g;
#pragma unroll
      for (int ks = 0; ks < 4; ++ks) {
        cx[g] = WMMA_BF16(Ax[ks], ldB(sWih1, nt * 4 + ks, lane), cx[g]);
        ch[g] = WMMA_BF16(Ah[ks], ldB(sWhh1, nt * 4 + ks, lane), ch[g]);
      }
    }
    __syncthreads();
#pragma unroll
    for (int r = 0; r < 8; ++r) {
      int m = r + mbase;
      float hp = H1[m * 132 + j];
      float rr = sigm(cx[0][r] + ch[0][r]);
      float zz = sigm(cx[1][r] + ch[1][r]);
      float nn = tanhf(cx[2][r] + rr * ch[2][r]);
      float hnew = (1.f - zz) * nn + zz * hp;
      H1[m * 132 + j] = hnew;
      acc[r] += hnew;
    }
    // No trailing barrier needed: next-iteration reads of Xt/H1 are separated
    // from this iteration's writes by the two barriers above in the next pass.
    __syncthreads();
  }

#pragma unroll
  for (int r = 0; r < 8; ++r) {
    int m = r + mbase;
    hmean[(size_t)(row0 + m) * 128 + j] = acc[r] * (1.0f / 256.0f);
  }
}

// ---------------------------------------------------------------------------
// GINE x2 + sum pool + classifier. One WG per batch. Aggregation in VALU over
// LDS; the [64x128]@[128x128] MLP via WMMA (4 tiles/wave).
// ---------------------------------------------------------------------------
__global__ void __launch_bounds__(256) gine_kernel(
    const float* __restrict__ hmean, const float* __restrict__ wg,
    const unsigned int* __restrict__ pNn0, const unsigned int* __restrict__ pNn1,
    const float* __restrict__ edge_W, const float* __restrict__ edge_b,
    const float* __restrict__ nn_b, const float* __restrict__ cls_W,
    const float* __restrict__ cls_b, float* __restrict__ out) {
  __shared__ float H[64 * 132];
  __shared__ float WM[64 * 64];
  __shared__ float red[256];
  int b = blockIdx.x, tid = threadIdx.x, lane = tid & 31, w = tid >> 5;
#pragma unroll
  for (int q = 0; q < 32; ++q) {
    int idx = tid + 256 * q;
    int n = idx >> 7, c = idx & 127;
    H[n * 132 + c] = hmean[(size_t)(b * 64 + n) * 128 + c];
  }
  for (int e = tid; e < 4096; e += 256) WM[e] = wg[(size_t)b * 4096 + e];
  __syncthreads();

  const int c = tid & 127, dst0 = tid >> 7;
  const int mt = w >> 1;
  for (int l = 0; l < 2; ++l) {
    const unsigned int* pNn = l ? pNn1 : pNn0;
    float eW = edge_W[l * 128 + c], eB = edge_b[l * 128 + c];
    float ag[32];
#pragma unroll
    for (int q = 0; q < 32; ++q) ag[q] = 0.f;
    for (int s = 0; s < 64; ++s) {
      float hv = H[s * 132 + c];
      const float* wrow = &WM[s * 64 + dst0];
#pragma unroll
      for (int q = 0; q < 32; ++q) {
        float wv = wrow[2 * q];
        float mm = fmaxf(hv + wv * eW + eB, 0.f);
        ag[q] += (wv > 0.f) ? mm : 0.f;
      }
    }
    __syncthreads();
#pragma unroll
    for (int q = 0; q < 32; ++q) H[(dst0 + 2 * q) * 132 + c] += ag[q];
    __syncthreads();
    v16bf A[4];
#pragma unroll
    for (int ks = 0; ks < 4; ++ks) A[ks] = makeA(&H[16 * mt * 132], lane, ks);
    __syncthreads();
    int mb = (lane & 16) ? 8 : 0;
#pragma unroll
    for (int i = 0; i < 4; ++i) {
      int nt = (w & 1) * 4 + i;
      float bias = nn_b[l * 128 + 16 * nt + (lane & 15)];
      v8f C;
#pragma unroll
      for (int r = 0; r < 8; ++r) C[r] = bias;
#pragma unroll
      for (int ks = 0; ks < 4; ++ks) C = WMMA_BF16(A[ks], ldB(pNn, nt * 4 + ks, lane), C);
#pragma unroll
      for (int r = 0; r < 8; ++r)
        H[(16 * mt + r + mb) * 132 + 16 * nt + (lane & 15)] = fmaxf(C[r], 0.f);
    }
    __syncthreads();
  }

  float p = 0.f;
#pragma unroll
  for (int q = 0; q < 32; ++q) {
    int idx = tid + 256 * q;
    int n = idx >> 7, cc = idx & 127;
    p += H[n * 132 + cc] * cls_W[cc];
  }
  red[tid] = p;
  __syncthreads();
  for (int s = 128; s > 0; s >>= 1) {
    if (tid < s) red[tid] += red[tid + s];
    __syncthreads();
  }
  if (tid == 0) out[b] = red[0] + cls_b[0];
}

// ---------------------------------------------------------------------------
// Launch
// ---------------------------------------------------------------------------
extern "C" void kernel_launch(void* const* d_in, const int* in_sizes, int n_in,
                              void* d_out, int out_size, void* d_ws, size_t ws_size,
                              hipStream_t stream) {
  (void)in_sizes; (void)n_in; (void)out_size; (void)ws_size;
  const float* x    = (const float*)d_in[0];
  const float* Wih0 = (const float*)d_in[1];
  const float* Whh0 = (const float*)d_in[2];
  const float* bih0 = (const float*)d_in[3];
  const float* bhh0 = (const float*)d_in[4];
  const float* Wih1 = (const float*)d_in[5];
  const float* Whh1 = (const float*)d_in[6];
  const float* bih1 = (const float*)d_in[7];
  const float* bhh1 = (const float*)d_in[8];
  const float* edgeW = (const float*)d_in[9];
  const float* edgeB = (const float*)d_in[10];
  const float* nnW   = (const float*)d_in[11];
  const float* nnB   = (const float*)d_in[12];
  const float* clsW  = (const float*)d_in[13];
  const float* clsB  = (const float*)d_in[14];

  char* ws = (char*)d_ws;
  unsigned int* pWih0 = (unsigned int*)(ws + 0);        // 96 frags * 1KB
  unsigned int* pWhh0 = (unsigned int*)(ws + 98304);
  unsigned int* pWih1 = (unsigned int*)(ws + 196608);
  unsigned int* pWhh1 = (unsigned int*)(ws + 294912);
  unsigned int* pNn0  = (unsigned int*)(ws + 393216);   // 32 frags * 1KB
  unsigned int* pNn1  = (unsigned int*)(ws + 425984);
  float* wgr          = (float*)(ws + 458752);          // 16*64*64*4
  float* hmean        = (float*)(ws + 720896);          // 1024*128*4

  pack_w_kernel<<<96, 32, 0, stream>>>(Wih0, pWih0, 128);
  pack_w_kernel<<<96, 32, 0, stream>>>(Whh0, pWhh0, 128);
  pack_w_kernel<<<96, 32, 0, stream>>>(Wih1, pWih1, 128);
  pack_w_kernel<<<96, 32, 0, stream>>>(Whh1, pWhh1, 128);
  pack_w_kernel<<<32, 32, 0, stream>>>(nnW, pNn0, 128);
  pack_w_kernel<<<32, 32, 0, stream>>>(nnW + 128 * 128, pNn1, 128);

  graph_build_kernel<<<16, 256, 0, stream>>>(x, wgr);
  gru_scan_kernel<<<64, 256, 0, stream>>>(x, pWih0, pWhh0, pWih1, pWhh1,
                                          bih0, bhh0, bih1, bhh1, hmean);
  gine_kernel<<<16, 256, 0, stream>>>(hmean, wgr, pNn0, pNn1, edgeW, edgeB,
                                      nnB, clsW, clsB, (float*)d_out);
}
